// MultiHeadAttention_21345987461791
// MI455X (gfx1250) — matrix operand order repaired and checked by reading, not compiled, not run on hardware
//
#include <hip/hip_runtime.h>
#include <hip/hip_bf16.h>

#define B_ 4
#define S_ 4096
#define HID_ 1024
#define NH_ 16
#define HD_ 64
#define ROWS_ (B_*S_)

typedef __attribute__((ext_vector_type(16))) __bf16        v16bf;
typedef __attribute__((ext_vector_type(8)))  float         v8f;
typedef __attribute__((ext_vector_type(8)))  unsigned int  v8u;
typedef unsigned short ushort_t;
typedef unsigned int   uint_t;

__device__ __forceinline__ float bf2f(ushort_t x) {
    uint_t u = ((uint_t)x) << 16;
    return __builtin_bit_cast(float, u);
}
__device__ __forceinline__ ushort_t f2bf(float f) {
    uint_t u = __builtin_bit_cast(uint_t, f);
    u += 0x7fffu + ((u >> 16) & 1u);           // round-to-nearest-even
    return (ushort_t)(u >> 16);
}

// ds_swizzle_b32 group-of-32 xor-lane reduction helper (pattern is an ICE)
#define SWZF(x, m) __builtin_bit_cast(float, __builtin_amdgcn_ds_swizzle( \
        __builtin_bit_cast(int, (x)), (((m) << 10) | 0x1f)))

// ---------------------------------------------------------------- fp32->bf16
__global__ __launch_bounds__(256) void convert_kernel(
    const float* __restrict__ in, ushort_t* __restrict__ out, int n) {
    int i = blockIdx.x * 256 + threadIdx.x;
    if (i < n) out[i] = f2bf(in[i]);
}

// fp32 W[k][n] -> bf16 Wt[n][k] (transposed so GEMM B-tiles are contiguous)
__global__ __launch_bounds__(256) void convert_wt_kernel(
    const float* __restrict__ in, ushort_t* __restrict__ out) {
    int i = blockIdx.x * 256 + threadIdx.x;       // over 1024*1024
    int k = i >> 10, n = i & 1023;
    out[(size_t)n * HID_ + k] = f2bf(in[i]);
}

// ---------------------------------------------------------------- GEMM
// Y[rows,1024] = X[rows,1024](bf16) @ W[1024,1024] + bias, f32 accum.
// Wt is W transposed: Wt[n][k]. 256 thr / 8 waves -> 128x128 tile.
// Waves: 4 (M) x 2 (N); each wave = 2 Mtiles x 4 Ntiles of 16x16 WMMA.
// K pipeline: async global->LDS (double buffered) overlapped with WMMA.

#define LSTR 20   // LDS row stride in dwords: 16 data + 4 pad (16B-aligned, conflict-free)

__device__ __forceinline__ void stage_async(const uint4* gbase, uint_t* ldsbase, int tid) {
    // copy 128 rows x 32 bf16 (4 uint4/row), global row stride 128 uint4
    unsigned lbase = (unsigned)(size_t)ldsbase;   // low 32 bits = LDS byte offset
    #pragma unroll
    for (int i = 0; i < 2; ++i) {
        int c = tid + i * 256;                    // 512 chunks of 16B
        int row = c >> 2, q = c & 3;
        unsigned long long ga = (unsigned long long)(gbase + (size_t)row * 128 + q);
        unsigned loff = lbase + (unsigned)(row * (LSTR * 4) + q * 16);
        asm volatile("global_load_async_to_lds_b128 %0, %1, off"
                     :: "v"(loff), "v"(ga) : "memory");
    }
}

template<bool OUT_BF16>
__global__ __launch_bounds__(256) void gemm_bf16_kernel(
    const ushort_t* __restrict__ X, const ushort_t* __restrict__ Wt,
    const float* __restrict__ bias, void* __restrict__ Y)
{
    __shared__ uint_t ldsA[2][128 * LSTR];
    __shared__ uint_t ldsB[2][128 * LSTR];
    const int tid  = threadIdx.x;
    const int lane = tid & 31, wave = tid >> 5;
    const int half = lane >> 4, l15 = lane & 15;
    const int m0 = blockIdx.y * 128, n0 = blockIdx.x * 128;
    const int wm = (wave & 3) * 32;     // wave M offset in tile
    const int wn = (wave >> 2) * 64;    // wave N offset in tile

    v8f acc[2][4];
    for (int nt = 0; nt < 4; ++nt) {
        float bv = bias[n0 + wn + nt * 16 + l15];
        for (int mt = 0; mt < 2; ++mt)
            for (int r = 0; r < 8; ++r) acc[mt][nt][r] = bv;
    }

    const uint4* Ab = (const uint4*)X  + (size_t)m0 * 128;  // 128 uint4 per row
    const uint4* Bb = (const uint4*)Wt + (size_t)n0 * 128;

    stage_async(Ab, ldsA[0], tid);
    stage_async(Bb, ldsB[0], tid);
    asm volatile("s_wait_asynccnt 0" ::: "memory");
    __syncthreads();

    for (int kt = 0; kt < HID_ / 32; ++kt) {
        const int cur = kt & 1;
        if (kt + 1 < HID_ / 32) {                 // prefetch next K tile (async DMA)
            stage_async(Ab + (size_t)(kt + 1) * 8, ldsA[cur ^ 1], tid);
            stage_async(Bb + (size_t)(kt + 1) * 8, ldsB[cur ^ 1], tid);
        }

        // fragments per CDNA5 16-bit WMMA layouts (two b128 LDS loads each)
        v8u afrag[2], bfrag[4];
        #pragma unroll
        for (int mt = 0; mt < 2; ++mt) {
            int row = wm + mt * 16 + l15;
            uint4 lo = *(const uint4*)&ldsA[cur][row * LSTR + half * 4];      // K 0..7|8..15
            uint4 hi = *(const uint4*)&ldsA[cur][row * LSTR + 8 + half * 4];  // K 16..23|24..31
            afrag[mt][0] = lo.x; afrag[mt][1] = lo.y; afrag[mt][2] = lo.z; afrag[mt][3] = lo.w;
            afrag[mt][4] = hi.x; afrag[mt][5] = hi.y; afrag[mt][6] = hi.z; afrag[mt][7] = hi.w;
        }
        #pragma unroll
        for (int nt = 0; nt < 4; ++nt) {
            int n = wn + nt * 16 + l15;
            uint4 lo = *(const uint4*)&ldsB[cur][n * LSTR + half * 4];        // K 0..7|8..15
            uint4 hi = *(const uint4*)&ldsB[cur][n * LSTR + 8 + half * 4];    // K 16..23|24..31
            bfrag[nt][0] = lo.x; bfrag[nt][1] = lo.y; bfrag[nt][2] = lo.z; bfrag[nt][3] = lo.w;
            bfrag[nt][4] = hi.x; bfrag[nt][5] = hi.y; bfrag[nt][6] = hi.z; bfrag[nt][7] = hi.w;
        }
        #pragma unroll
        for (int mt = 0; mt < 2; ++mt)
            #pragma unroll
            for (int nt = 0; nt < 4; ++nt)
                acc[mt][nt] = __builtin_amdgcn_wmma_f32_16x16x32_bf16(
                    false, __builtin_bit_cast(v16bf, afrag[mt]),
                    false, __builtin_bit_cast(v16bf, bfrag[nt]),
                    (short)0, acc[mt][nt], false, false);

        asm volatile("s_wait_asynccnt 0" ::: "memory");  // next tile landed in LDS
        __syncthreads();
    }

    #pragma unroll
    for (int mt = 0; mt < 2; ++mt)
        for (int nt = 0; nt < 4; ++nt)
            for (int r = 0; r < 8; ++r) {
                int row = m0 + wm + mt * 16 + half * 8 + r;
                int col = n0 + wn + nt * 16 + l15;
                float val = acc[mt][nt][r];
                if (OUT_BF16) ((ushort_t*)Y)[(size_t)row * HID_ + col] = f2bf(val);
                else          ((float*)Y)  [(size_t)row * HID_ + col] = val;
            }
}

// ---------------------------------------------------------------- attention
// Per token: S = (Q Kt)/8 over heads (16x16, K=64 via 2 WMMAs),
// softmax over key-heads via ds_swizzle xor-reductions, O = P V on VALU.
__global__ __launch_bounds__(256) void attn_kernel(
    const ushort_t* __restrict__ qh, const ushort_t* __restrict__ kh,
    const ushort_t* __restrict__ vh, ushort_t* __restrict__ out)
{
    __shared__ float Plds[8][16][17];
    const int tid = threadIdx.x, lane = tid & 31, wave = tid >> 5;
    const int half = lane >> 4, l15 = lane & 15;
    const int token = blockIdx.x * 8 + wave;

    const uint_t* qdw = (const uint_t*)(qh + (size_t)token * HID_);
    const uint_t* kdw = (const uint_t*)(kh + (size_t)token * HID_);

    v8f s;
    for (int r = 0; r < 8; ++r) s[r] = 0.0f;

    #pragma unroll
    for (int kt = 0; kt < 2; ++kt) {
        v8u a, b;
        #pragma unroll
        for (int v = 0; v < 8; ++v) {
            // A: row h=l15, K=d contiguous pairs in head-major Q
            a[v] = qdw[l15 * 32 + kt * 16 + (v >> 2) * 8 + half * 4 + (v & 3)];
            // B = K^T: col t=l15, K=d contiguous pairs in head-major K
            b[v] = kdw[l15 * 32 + kt * 16 + (v >> 2) * 8 + half * 4 + (v & 3)];
        }
        s = __builtin_amdgcn_wmma_f32_16x16x32_bf16(
                false, __builtin_bit_cast(v16bf, a),
                false, __builtin_bit_cast(v16bf, b),
                (short)0, s, false, false);
    }

    const float scale = 0.125f;               // 1/sqrt(64)
    #pragma unroll
    for (int r = 0; r < 8; ++r) {             // row h = r + half*8, col t = l15
        float x = s[r] * scale;
        float mx = x;
        mx = fmaxf(mx, SWZF(mx, 1));
        mx = fmaxf(mx, SWZF(mx, 2));
        mx = fmaxf(mx, SWZF(mx, 4));
        mx = fmaxf(mx, SWZF(mx, 8));
        float p = __expf(x - mx);
        float sum = p;
        sum += SWZF(sum, 1);
        sum += SWZF(sum, 2);
        sum += SWZF(sum, 4);
        sum += SWZF(sum, 8);
        // softmax normalization: HW v_rcp_f32 (~1 ulp) instead of IEEE div chain
        Plds[wave][r + half * 8][l15] = p * __builtin_amdgcn_rcpf(sum);
    }
    // same-wave DS ops are in order; P fully written before reads below

    const ushort_t* vt = vh + (size_t)token * HID_;
    float v0[16], v1[16];
    #pragma unroll
    for (int t = 0; t < 16; ++t) {
        v0[t] = bf2f(vt[t * 64 + lane]);
        v1[t] = bf2f(vt[t * 64 + lane + 32]);
    }
    ushort_t* ot = out + (size_t)token * HID_;
    #pragma unroll
    for (int h = 0; h < 16; ++h) {
        float o0 = 0.0f, o1 = 0.0f;
        #pragma unroll
        for (int t = 0; t < 16; ++t) {
            float p = Plds[wave][h][t];
            o0 += p * v0[t];
            o1 += p * v1[t];
        }
        ot[h * 64 + lane]      = f2bf(o0);
        ot[h * 64 + lane + 32] = f2bf(o1);
    }
}

// ---------------------------------------------------------------- layernorm
// Faithful to reference: var = sum((x-mu)^2)/(HID-1); y = w*(x-mu)/(var+eps)+b
__global__ __launch_bounds__(256) void ln_kernel(
    const float* __restrict__ X, const float* __restrict__ w,
    const float* __restrict__ b, float* __restrict__ Y)
{
    __shared__ float red[8];
    const int row = blockIdx.x, tid = threadIdx.x;
    const int lane = tid & 31, wave = tid >> 5;
    const float* x = X + (size_t)row * HID_;
    float v[4];
    #pragma unroll
    for (int i = 0; i < 4; ++i) v[i] = x[tid + i * 256];

    float s = v[0] + v[1] + v[2] + v[3];
    s += SWZF(s, 1); s += SWZF(s, 2); s += SWZF(s, 4); s += SWZF(s, 8); s += SWZF(s, 16);
    if (lane == 0) red[wave] = s;
    __syncthreads();
    float tot = 0.0f;
    for (int i = 0; i < 8; ++i) tot += red[i];
    float mu = tot * (1.0f / 1024.0f);
    __syncthreads();                            // red[] reuse barrier

    float d = 0.0f;
    #pragma unroll
    for (int i = 0; i < 4; ++i) { float t = v[i] - mu; d += t * t; }
    d += SWZF(d, 1); d += SWZF(d, 2); d += SWZF(d, 4); d += SWZF(d, 8); d += SWZF(d, 16);
    if (lane == 0) red[wave] = d;
    __syncthreads();
    float tot2 = 0.0f;
    for (int i = 0; i < 8; ++i) tot2 += red[i];
    float var = tot2 * (1.0f / 1023.0f);
    float inv = __builtin_amdgcn_rcpf(var + 1e-5f);
    #pragma unroll
    for (int i = 0; i < 4; ++i) {
        int c = tid + i * 256;
        Y[(size_t)row * HID_ + c] = w[c] * (v[i] - mu) * inv + b[c];
    }
}

// ---------------------------------------------------------------- launch
extern "C" void kernel_launch(void* const* d_in, const int* in_sizes, int n_in,
                              void* d_out, int out_size, void* d_ws, size_t ws_size,
                              hipStream_t stream) {
    (void)in_sizes; (void)n_in; (void)out_size; (void)ws_size;
    const float* q   = (const float*)d_in[0];
    const float* k   = (const float*)d_in[1];
    const float* v   = (const float*)d_in[2];
    const float* Wq  = (const float*)d_in[3];
    const float* bq  = (const float*)d_in[4];
    const float* Wk  = (const float*)d_in[5];
    const float* bk  = (const float*)d_in[6];
    const float* Wv  = (const float*)d_in[7];
    const float* bv  = (const float*)d_in[8];
    const float* Wo  = (const float*)d_in[9];
    const float* bo  = (const float*)d_in[10];
    const float* lnw = (const float*)d_in[11];
    const float* lnb = (const float*)d_in[12];

    // workspace layout (~200 MB total)
    char* ws = (char*)d_ws;
    const size_t SZ_W = (size_t)HID_ * HID_ * 2;     // 2 MB per weight
    const size_t SZ_X = (size_t)ROWS_ * HID_ * 2;    // 32 MB per activation
    ushort_t* wq_bf  = (ushort_t*)(ws);                        // transposed bf16
    ushort_t* wk_bf  = (ushort_t*)(ws + SZ_W);
    ushort_t* wv_bf  = (ushort_t*)(ws + 2 * SZ_W);
    ushort_t* wo_bf  = (ushort_t*)(ws + 3 * SZ_W);
    ushort_t* buf_bf = (ushort_t*)(ws + 4 * SZ_W);             // reused staging
    ushort_t* qh_bf  = (ushort_t*)(ws + 4 * SZ_W + SZ_X);
    ushort_t* kh_bf  = (ushort_t*)(ws + 4 * SZ_W + 2 * SZ_X);
    ushort_t* vh_bf  = (ushort_t*)(ws + 4 * SZ_W + 3 * SZ_X);
    float*    proj   = (float*)   (ws + 4 * SZ_W + 4 * SZ_X);  // 64 MB f32

    const int NW = HID_ * HID_;
    const int NX = ROWS_ * HID_;
    dim3 cb(256);
    dim3 ggrid(HID_ / 128, ROWS_ / 128);   // (8, 128)

    convert_wt_kernel<<<NW / 256, cb, 0, stream>>>(Wq, wq_bf);
    convert_wt_kernel<<<NW / 256, cb, 0, stream>>>(Wk, wk_bf);
    convert_wt_kernel<<<NW / 256, cb, 0, stream>>>(Wv, wv_bf);
    convert_wt_kernel<<<NW / 256, cb, 0, stream>>>(Wo, wo_bf);

    convert_kernel<<<NX / 256, cb, 0, stream>>>(q, buf_bf, NX);
    gemm_bf16_kernel<true><<<ggrid, cb, 0, stream>>>(buf_bf, wq_bf, bq, qh_bf);
    convert_kernel<<<NX / 256, cb, 0, stream>>>(k, buf_bf, NX);
    gemm_bf16_kernel<true><<<ggrid, cb, 0, stream>>>(buf_bf, wk_bf, bk, kh_bf);
    convert_kernel<<<NX / 256, cb, 0, stream>>>(v, buf_bf, NX);
    gemm_bf16_kernel<true><<<ggrid, cb, 0, stream>>>(buf_bf, wv_bf, bv, vh_bf);

    attn_kernel<<<ROWS_ / 8, cb, 0, stream>>>(qh_bf, kh_bf, vh_bf, buf_bf);

    gemm_bf16_kernel<false><<<ggrid, cb, 0, stream>>>(buf_bf, wo_bf, bo, proj);
    ln_kernel<<<ROWS_, cb, 0, stream>>>(proj, lnw, lnb, (float*)d_out);
}
